// IlseBagModel_10127532884279
// MI455X (gfx1250) — compile-verified, matrix operand order checked by reading, not run on hardware
//
#include <hip/hip_runtime.h>
#include <hip/hip_bf16.h>
#include <math.h>

typedef _Float16 v16h __attribute__((ext_vector_type(16)));
typedef _Float16 v8h  __attribute__((ext_vector_type(8)));
typedef float    v8f  __attribute__((ext_vector_type(8)));

#define N_INST 1024
#define NF     512
#define HW     196      // 14*14
#define L_DIM  128
#define C_DIM  2
#define NBAGS  128
#define FC     32       // f-rows staged per LDS tile in k_sal
#define TILE_CHUNKS 1568   // FC*HW*4 / 16 = 25088 B / 16 B

// ---------------------------------------------------------------------------
// Kernel 0: convert Wv/Wu (f32 [128,512]) -> f16 copies for WMMA B operands
// ---------------------------------------------------------------------------
__global__ __launch_bounds__(256) void k_cvt(const float* __restrict__ Wv,
                                             const float* __restrict__ Wu,
                                             _Float16* __restrict__ Wv16,
                                             _Float16* __restrict__ Wu16) {
    int i = blockIdx.x * 256 + threadIdx.x;   // grid covers L_DIM*NF = 65536
    Wv16[i] = (_Float16)Wv[i];
    Wu16[i] = (_Float16)Wu[i];
}

// ---------------------------------------------------------------------------
// Async tile DMA: global -> LDS via GLOBAL_LOAD_ASYNC_TO_LDS_B128 (ASYNCcnt).
// 1568 chunks of 16 B: 6 rounds x 256 threads + 32 remainder chunks issued by
// wave 0 only. Per-wave issue count: wave0 = 7, waves 1..7 = 6.
// ---------------------------------------------------------------------------
__device__ __forceinline__ void issue_tile(const float* __restrict__ src,
                                           unsigned lds_base, int t) {
#pragma unroll
    for (int r = 0; r < 6; ++r) {
        int c = t + r * 256;
        unsigned long long ga = (unsigned long long)(size_t)src +
                                (unsigned long long)c * 16ull;
        unsigned la = lds_base + (unsigned)c * 16u;
        asm volatile("global_load_async_to_lds_b128 %0, %1, off"
                     :: "v"(la), "v"(ga) : "memory");
    }
    if (t < 32) {  // wave 0 (wave-uniform branch)
        int c = 1536 + t;
        unsigned long long ga = (unsigned long long)(size_t)src +
                                (unsigned long long)c * 16ull;
        unsigned la = lds_base + (unsigned)c * 16u;
        asm volatile("global_load_async_to_lds_b128 %0, %1, off"
                     :: "v"(la), "v"(ga) : "memory");
    }
}

__device__ __forceinline__ void wait_tile(int wave, bool last) {
    if (last) {
        asm volatile("s_wait_asynccnt 0x0" ::: "memory");
    } else if (wave == 0) {
        asm volatile("s_wait_asynccnt 0x7" ::: "memory");
    } else {
        asm volatile("s_wait_asynccnt 0x6" ::: "memory");
    }
}

// ---------------------------------------------------------------------------
// Kernel 1: per-instance streaming pass over h[n] (the 411 MB tensor).
//   - double-buffered async global->LDS tile DMA (ASYNCcnt)
//   - 1x1 conv (C=2) accumulated in registers (thread t = spatial pos)
//   - spatial max per f via wave shfl_xor reduction -> v[n,f] (f16)
//   - tail: top-3 of conv logits, sigmoid, mean -> yhat_instance[n,c]
// ---------------------------------------------------------------------------
__global__ __launch_bounds__(256) void k_sal(const float* __restrict__ h,
                                             const float* __restrict__ conv_w,
                                             _Float16* __restrict__ vf16,
                                             float* __restrict__ yinst) {
    __shared__ __align__(16) float tile[2][FC][HW];  // 2 x 25088 B
    __shared__ float cw[C_DIM][NF];                  // 4096 B
    __shared__ float sal[C_DIM][HW];                 // 1568 B

    const int t    = threadIdx.x;
    const int lane = t & 31;
    const int wave = t >> 5;
    const int n    = blockIdx.x;
    const float* hn = h + (size_t)n * NF * HW;

    const unsigned lds0 = (unsigned)(size_t)(void*)&tile[0][0][0];
    const unsigned lds1 = (unsigned)(size_t)(void*)&tile[1][0][0];

    for (int i = t; i < C_DIM * NF; i += 256) cw[i >> 9][i & (NF - 1)] = conv_w[i];

    // kick off tile 0
    issue_tile(hn, lds0, t);

    float acc0 = 0.f, acc1 = 0.f;
    __syncthreads();   // cw visible (async tile still in flight)

    for (int f0 = 0; f0 < NF; f0 += FC) {
        const int  cur      = (f0 / FC) & 1;
        const bool haveNext = (f0 + FC) < NF;

        // issue next tile into the other buffer (drained at end of prev iter)
        if (haveNext)
            issue_tile(hn + (size_t)(f0 + FC) * HW, cur ? lds0 : lds1, t);

        // warm L2 for tile after next
        if (f0 + 2 * FC < NF) {
            const float* nxt2 = hn + (size_t)(f0 + 2 * FC) * HW;
            for (int i = t * 32; i < FC * HW; i += 256 * 32)
                __builtin_prefetch(nxt2 + i, 0, 1);
        }

        // retire current tile's async ops (per-wave), then make tile visible
        wait_tile(wave, !haveNext);
        __syncthreads();

        const float (*tl)[HW] = tile[cur];

        // conv accumulation: thread t owns spatial position t
        if (t < HW) {
#pragma unroll 8
            for (int fl = 0; fl < FC; ++fl) {
                float x = tl[fl][t];
                acc0 = fmaf(cw[0][f0 + fl], x, acc0);
                acc1 = fmaf(cw[1][f0 + fl], x, acc1);
            }
        }

        // spatial max per f-row: 8 waves x 4 rows
        for (int r = wave * 4; r < wave * 4 + 4; ++r) {
            float m = -3.4e38f;
            for (int j = lane; j < HW; j += 32) m = fmaxf(m, tl[r][j]);
            for (int s = 16; s > 0; s >>= 1) m = fmaxf(m, __shfl_xor(m, s, 32));
            if (lane == 0) vf16[(size_t)n * NF + f0 + r] = (_Float16)m;
        }
        __syncthreads();   // all waves done with tile[cur] before it is reused
    }

    if (t < HW) { sal[0][t] = acc0; sal[1][t] = acc1; }
    __syncthreads();

    // top-3 (sigmoid monotonic -> rank on raw logits), mean of sigmoid
    if (t < C_DIM) {
        float t1 = -3.4e38f, t2 = -3.4e38f, t3 = -3.4e38f;
        for (int i = 0; i < HW; ++i) {
            float x = sal[t][i];
            if (x > t1)      { t3 = t2; t2 = t1; t1 = x; }
            else if (x > t2) { t3 = t2; t2 = x; }
            else if (x > t3) { t3 = x; }
        }
        float s = (1.f / (1.f + __expf(-t1)) +
                   1.f / (1.f + __expf(-t2)) +
                   1.f / (1.f + __expf(-t3))) * (1.f / 3.f);
        yinst[n * C_DIM + t] = s;
    }
}

// ---------------------------------------------------------------------------
// Kernel 2: gated attention via WMMA.
//   Two f32-accum f16 WMMA chains per wave (v@Wv.T and v@Wu.T share A-frag),
//   then gate (tanh*sigmoid), project onto Ww (C=2), LDS-atomic row reduce.
// Block = 256 threads = 8 waves; block b -> rows [16b,16b+16); wave w -> cols
// [16w,16w+16). K loop over NF in steps of 32.
// ---------------------------------------------------------------------------
__global__ __launch_bounds__(256) void k_attn(const _Float16* __restrict__ vf16,
                                              const _Float16* __restrict__ Wv16,
                                              const _Float16* __restrict__ Wu16,
                                              const float* __restrict__ bv,
                                              const float* __restrict__ bu,
                                              const float* __restrict__ Ww,
                                              const float* __restrict__ bw,
                                              float* __restrict__ A) {
    __shared__ float Apart[16][C_DIM];

    const int t    = threadIdx.x;
    const int lane = t & 31;
    const int wave = t >> 5;
    const int m0   = blockIdx.x * 16;
    const int n0   = wave * 16;
    const int half = lane >> 4;     // 0: K 0-7/16-23, 1: K 8-15/24-31
    const int l15  = lane & 15;

    if (t < 32) ((float*)Apart)[t] = 0.f;
    __syncthreads();

    const _Float16* ap  = vf16 + (size_t)(m0 + l15) * NF + half * 8;
    const _Float16* bvp = Wv16 + (size_t)(n0 + l15) * NF + half * 8;
    const _Float16* bup = Wu16 + (size_t)(n0 + l15) * NF + half * 8;

    v8f cv = {};
    v8f cu = {};
    for (int k0 = 0; k0 < NF; k0 += 32) {
        v8h alo = *(const v8h*)(ap + k0);
        v8h ahi = *(const v8h*)(ap + k0 + 16);
        v16h a = __builtin_shufflevector(alo, ahi, 0, 1, 2, 3, 4, 5, 6, 7,
                                         8, 9, 10, 11, 12, 13, 14, 15);
        v8h vlo = *(const v8h*)(bvp + k0);
        v8h vhi = *(const v8h*)(bvp + k0 + 16);
        v16h bV = __builtin_shufflevector(vlo, vhi, 0, 1, 2, 3, 4, 5, 6, 7,
                                          8, 9, 10, 11, 12, 13, 14, 15);
        v8h ulo = *(const v8h*)(bup + k0);
        v8h uhi = *(const v8h*)(bup + k0 + 16);
        v16h bU = __builtin_shufflevector(ulo, uhi, 0, 1, 2, 3, 4, 5, 6, 7,
                                          8, 9, 10, 11, 12, 13, 14, 15);
        cv = __builtin_amdgcn_wmma_f32_16x16x32_f16(false, a, false, bV,
                                                    (short)0, cv, false, false);
        cu = __builtin_amdgcn_wmma_f32_16x16x32_f16(false, a, false, bU,
                                                    (short)0, cu, false, false);
    }

    // epilogue: C/D layout -> lane holds column n = n0+l15, rows j + 8*half
    const int ncol = n0 + l15;
    const float biasv = bv[ncol];
    const float biasu = bu[ncol];
    const float w0 = Ww[ncol];
    const float w1 = Ww[L_DIM + ncol];
#pragma unroll
    for (int j = 0; j < 8; ++j) {
        int mrow = j + half * 8;
        float gv = tanhf(cv[j] + biasv);
        float gu = 1.f / (1.f + __expf(-(cu[j] + biasu)));
        float g  = gv * gu;
        atomicAdd(&Apart[mrow][0], g * w0);
        atomicAdd(&Apart[mrow][1], g * w1);
    }
    __syncthreads();

    if (t < 32) {
        int row = t >> 1, cls = t & 1;
        A[(m0 + row) * C_DIM + cls] = Apart[row][cls] + bw[cls];
    }
}

// ---------------------------------------------------------------------------
// Kernel 3: per-bag segment softmax over A + attention-weighted sum of
// yhat_instance. One block per bag; masked tree reductions in LDS.
// ---------------------------------------------------------------------------
__global__ __launch_bounds__(256) void k_bags(const float* __restrict__ A,
                                              const float* __restrict__ yinst,
                                              const int* __restrict__ ids,
                                              float* __restrict__ out) {
    __shared__ float red[256];
    const int t   = threadIdx.x;
    const int bag = blockIdx.x;

    for (int c = 0; c < C_DIM; ++c) {
        float lm = -3.4e38f;
        for (int i = t; i < N_INST; i += 256)
            if (ids[i] == bag) lm = fmaxf(lm, A[i * C_DIM + c]);
        red[t] = lm;
        __syncthreads();
        for (int s = 128; s > 0; s >>= 1) {
            if (t < s) red[t] = fmaxf(red[t], red[t + s]);
            __syncthreads();
        }
        const float m = red[0];
        __syncthreads();

        float lsum = 0.f, lwy = 0.f;
        for (int i = t; i < N_INST; i += 256) {
            if (ids[i] == bag) {
                float w = __expf(A[i * C_DIM + c] - m);
                lsum += w;
                lwy  += w * yinst[i * C_DIM + c];
            }
        }
        red[t] = lsum;
        __syncthreads();
        for (int s = 128; s > 0; s >>= 1) {
            if (t < s) red[t] += red[t + s];
            __syncthreads();
        }
        const float denom = red[0];
        __syncthreads();

        red[t] = lwy;
        __syncthreads();
        for (int s = 128; s > 0; s >>= 1) {
            if (t < s) red[t] += red[t + s];
            __syncthreads();
        }
        if (t == 0) out[bag * C_DIM + c] = red[0] / denom;
        __syncthreads();
    }
}

// ---------------------------------------------------------------------------
extern "C" void kernel_launch(void* const* d_in, const int* in_sizes, int n_in,
                              void* d_out, int out_size, void* d_ws, size_t ws_size,
                              hipStream_t stream) {
    const float* h      = (const float*)d_in[0];
    const int*   ids    = (const int*)d_in[1];
    const float* conv_w = (const float*)d_in[2];
    const float* Wv     = (const float*)d_in[3];
    const float* bv     = (const float*)d_in[4];
    const float* Wu     = (const float*)d_in[5];
    const float* bu     = (const float*)d_in[6];
    const float* Ww     = (const float*)d_in[7];
    const float* bw     = (const float*)d_in[8];
    float* out = (float*)d_out;

    char* ws = (char*)d_ws;
    _Float16* vf16  = (_Float16*)(ws);                      // 1024*512*2   = 1,048,576 B
    _Float16* Wv16  = (_Float16*)(ws + 1048576);            // 128*512*2    =   131,072 B
    _Float16* Wu16  = (_Float16*)(ws + 1048576 + 131072);   //                 131,072 B
    float*    yinst = (float*)(ws + 1310720);               // 1024*2*4     =     8,192 B
    float*    Amat  = (float*)(ws + 1318912);               // 1024*2*4     =     8,192 B

    k_cvt <<<(L_DIM * NF) / 256, 256, 0, stream>>>(Wv, Wu, Wv16, Wu16);
    k_sal <<<N_INST, 256, 0, stream>>>(h, conv_w, vf16, yinst);
    k_attn<<<N_INST / 16, 256, 0, stream>>>(vf16, Wv16, Wu16, bv, bu, Ww, bw, Amat);
    k_bags<<<NBAGS, 256, 0, stream>>>(Amat, yinst, ids, out);
}
